// CCL_2954937499678
// MI455X (gfx1250) — matrix operand order repaired and checked by reading, not compiled, not run on hardware
//
#include <hip/hip_runtime.h>
#include <math.h>

#define NN 50000
#define EE 400000
#define DIN 256
#define DH 256
#define DOUT 64
#define EPS 1e-5f

typedef __attribute__((ext_vector_type(16))) __bf16 bf16x16;
typedef __attribute__((ext_vector_type(8)))  __bf16 bf16x8;
typedef __attribute__((ext_vector_type(8)))  float  f32x8;

// ---------------- utility ----------------
__global__ void k_zero(float* p, long n) {
    long i = (long)blockIdx.x * blockDim.x + threadIdx.x;
    long stride = (long)gridDim.x * blockDim.x;
    for (; i < n; i += stride) p[i] = 0.0f;
}

// Pack weight_gcn (DIN x DH, row-major, B[k][n]=W[k*DH+n]) into wave32 B-fragment order:
// idx = ((tileN*8 + kstep)*32 + lane)*16 + e ; k = kstep*32 + (lane>=16?8:0) + (e&7) + (e>=8?16:0) ; col = tileN*16 + (lane&15)
__global__ void k_pack_w(const float* __restrict__ W, __bf16* __restrict__ out) {
    int i = blockIdx.x * blockDim.x + threadIdx.x;   // [0, DIN*DH)
    int e = i & 15, lane = (i >> 4) & 31, ks = (i >> 9) & 7, tn = i >> 12;
    int k = ks * 32 + ((lane >> 4) << 3) + (e & 7) + ((e >> 3) << 4);
    int col = tn * 16 + (lane & 15);
    out[i] = (__bf16)W[k * DH + col];
}

// Pack prototypes (DOUT x DH): B[k][n] = P[n*DH + k]
__global__ void k_pack_p(const float* __restrict__ P, __bf16* __restrict__ out) {
    int i = blockIdx.x * blockDim.x + threadIdx.x;   // [0, DH*DOUT)
    int e = i & 15, lane = (i >> 4) & 31, ks = (i >> 9) & 7, tn = i >> 12;
    int k = ks * 32 + ((lane >> 4) << 3) + (e & 7) + ((e >> 3) << 4);
    int col = tn * 16 + (lane & 15);
    out[i] = (__bf16)P[col * DH + k];
}

// ---------------- SPMM scatter: out[src] += w * in[dst] ----------------
__global__ void k_spmm(const int* __restrict__ src, const int* __restrict__ dst,
                       const float* __restrict__ w, const float* __restrict__ in,
                       float* __restrict__ out) {
    int e = blockIdx.x;
    int f = threadIdx.x;
    float wv = w[e];
    float v = wv * in[(size_t)dst[e] * DIN + f];
    atomicAdd(&out[(size_t)src[e] * DIN + f], v);
}

// ---------------- BN1 stats over xm = (x + c1 + c2)/3 ; one block per feature ----------------
__global__ void k_bn1_stats(const float* __restrict__ x, const float* __restrict__ c1,
                            const float* __restrict__ c2, const float* __restrict__ gamma,
                            const float* __restrict__ beta,
                            float* __restrict__ scale, float* __restrict__ shift) {
    __shared__ float ss[256], s2[256];
    int f = blockIdx.x, tid = threadIdx.x;
    float a = 0.f, b = 0.f;
    for (int r = tid; r < NN; r += 256) {
        size_t idx = (size_t)r * DIN + f;
        float v = (x[idx] + c1[idx] + c2[idx]) * (1.0f / 3.0f);
        a += v; b += v * v;
    }
    ss[tid] = a; s2[tid] = b; __syncthreads();
    for (int o = 128; o > 0; o >>= 1) {
        if (tid < o) { ss[tid] += ss[tid + o]; s2[tid] += s2[tid + o]; }
        __syncthreads();
    }
    if (tid == 0) {
        float mu = ss[0] / NN;
        float var = s2[0] / NN - mu * mu;
        float sc = gamma[f] * rsqrtf(var + EPS);
        scale[f] = sc;
        shift[f] = beta[f] - mu * sc;
    }
}

// ---------------- BN2 stats over h ----------------
__global__ void k_bn2_stats(const float* __restrict__ h, const float* __restrict__ gamma,
                            const float* __restrict__ beta,
                            float* __restrict__ scale, float* __restrict__ shift) {
    __shared__ float ss[256], s2[256];
    int f = blockIdx.x, tid = threadIdx.x;
    float a = 0.f, b = 0.f;
    for (int r = tid; r < NN; r += 256) {
        float v = h[(size_t)r * DH + f];
        a += v; b += v * v;
    }
    ss[tid] = a; s2[tid] = b; __syncthreads();
    for (int o = 128; o > 0; o >>= 1) {
        if (tid < o) { ss[tid] += ss[tid + o]; s2[tid] += s2[tid + o]; }
        __syncthreads();
    }
    if (tid == 0) {
        float mu = ss[0] / NN;
        float var = s2[0] / NN - mu * mu;
        float sc = gamma[f] * rsqrtf(var + EPS);
        scale[f] = sc;
        shift[f] = beta[f] - mu * sc;
    }
}

// ---------------- fused BN1 affine -> bf16 feature matrix ----------------
__global__ void k_xm_bf16(const float* __restrict__ x, const float* __restrict__ c1,
                          const float* __restrict__ c2, const float* __restrict__ scale,
                          const float* __restrict__ shift, __bf16* __restrict__ out) {
    long i = (long)blockIdx.x * blockDim.x + threadIdx.x;
    long n = (long)NN * DIN;
    long stride = (long)gridDim.x * blockDim.x;
    for (; i < n; i += stride) {
        int f = (int)(i & (DIN - 1));
        float v = (x[i] + c1[i] + c2[i]) * (1.0f / 3.0f);
        out[i] = (__bf16)(scale[f] * v + shift[f]);
    }
}

// ---------------- GEMM1: h = relu(xm_bf16 @ Wpack + bias) ----------------
// One wave computes a 16x64 strip: 4 N-tiles sharing one A fragment per k-step.
__global__ void k_gemm1(const __bf16* __restrict__ A, const __bf16* __restrict__ Bp,
                        const float* __restrict__ bias, float* __restrict__ h) {
    int wave = (blockIdx.x << 3) + (threadIdx.x >> 5);   // 8 waves/block
    if (wave >= (NN / 16) * ((DH / 16) / 4)) return;     // 3125*4 = 12500 waves, uniform exit
    int lane = threadIdx.x & 31;
    int tileM = wave >> 2;
    int nb = (wave & 3) * 4;          // first of 4 consecutive N-tiles
    int l15 = lane & 15, mh = lane >> 4;
    int row = tileM * 16 + l15;       // A row for this lane
    int kh = mh * 8;

    f32x8 acc[4] = {};
    const __bf16* arow = A + (size_t)row * DIN;
    __builtin_prefetch(arow, 0, 0);   // global_prefetch_b8 on the A stream
    #pragma unroll
    for (int ks = 0; ks < DIN / 32; ++ks) {
        int k0 = ks * 32;
        bf16x8 alo = *(const bf16x8*)(arow + k0 + kh);
        bf16x8 ahi = *(const bf16x8*)(arow + k0 + 16 + kh);
        bf16x16 a;
        #pragma unroll
        for (int i = 0; i < 8; ++i) { a[i] = alo[i]; a[i + 8] = ahi[i]; }
        #pragma unroll
        for (int t = 0; t < 4; ++t) {
            bf16x16 b = *(const bf16x16*)(Bp + (((size_t)((nb + t) * 8 + ks) * 32 + lane) << 4));
            acc[t] = __builtin_amdgcn_wmma_f32_16x16x32_bf16(false, a, false, b,
                                                             (short)0, acc[t], false, false);
        }
    }
    #pragma unroll
    for (int t = 0; t < 4; ++t) {
        int col = (nb + t) * 16 + l15;
        float bv = bias[col];
        #pragma unroll
        for (int j = 0; j < 8; ++j) {
            int m = tileM * 16 + j + mh * 8;
            float v = acc[t][j] + bv;
            h[(size_t)m * DH + col] = v > 0.f ? v : 0.f;
        }
    }
}

// ---------------- GEMM2: z_pre = bn2(h) @ Ppack + bias_pre ----------------
// One wave computes the full 16x64 output strip (DOUT = 4 N-tiles); BN affine
// + f32->bf16 conversion of the A fragment is done once per k-step.
__global__ void k_gemm2(const float* __restrict__ H, const __bf16* __restrict__ Bp,
                        const float* __restrict__ scale, const float* __restrict__ shift,
                        const float* __restrict__ bias, float* __restrict__ z) {
    int wave = (blockIdx.x << 3) + (threadIdx.x >> 5);
    if (wave >= NN / 16) return;                         // 3125 waves, uniform per-wave exit
    int lane = threadIdx.x & 31;
    int tileM = wave;
    int l15 = lane & 15, mh = lane >> 4;
    int row = tileM * 16 + l15;
    int kh = mh * 8;

    f32x8 acc[4] = {};
    const float* hrow = H + (size_t)row * DH;
    __builtin_prefetch(hrow, 0, 0);
    #pragma unroll
    for (int ks = 0; ks < DH / 32; ++ks) {
        int k0 = ks * 32;
        bf16x16 a;
        #pragma unroll
        for (int i = 0; i < 8; ++i) {
            int ka = k0 + kh + i;
            int kb = k0 + 16 + kh + i;
            a[i]     = (__bf16)(hrow[ka] * scale[ka] + shift[ka]);
            a[i + 8] = (__bf16)(hrow[kb] * scale[kb] + shift[kb]);
        }
        #pragma unroll
        for (int t = 0; t < 4; ++t) {
            bf16x16 b = *(const bf16x16*)(Bp + (((size_t)(t * 8 + ks) * 32 + lane) << 4));
            acc[t] = __builtin_amdgcn_wmma_f32_16x16x32_bf16(false, a, false, b,
                                                             (short)0, acc[t], false, false);
        }
    }
    #pragma unroll
    for (int t = 0; t < 4; ++t) {
        int col = t * 16 + l15;
        float bv = bias[col];
        #pragma unroll
        for (int j = 0; j < 8; ++j) {
            int m = tileM * 16 + j + mh * 8;
            z[(size_t)m * DOUT + col] = acc[t][j] + bv;
        }
    }
}

// ---------------- row-wise log_softmax(z / t_p), in place ----------------
__global__ void k_logsoftmax(float* __restrict__ z, const float* __restrict__ t_p) {
    __shared__ float sh[DOUT];
    int row = blockIdx.x, tid = threadIdx.x;
    float inv = 1.0f / t_p[0];
    float v = z[(size_t)row * DOUT + tid] * inv;
    sh[tid] = v; __syncthreads();
    for (int o = 32; o > 0; o >>= 1) { if (tid < o) sh[tid] = fmaxf(sh[tid], sh[tid + o]); __syncthreads(); }
    float m = sh[0]; __syncthreads();
    sh[tid] = __expf(v - m); __syncthreads();
    for (int o = 32; o > 0; o >>= 1) { if (tid < o) sh[tid] += sh[tid + o]; __syncthreads(); }
    float l = __logf(sh[0]);
    z[(size_t)row * DOUT + tid] = v - m - l;
}

extern "C" void kernel_launch(void* const* d_in, const int* in_sizes, int n_in,
                              void* d_out, int out_size, void* d_ws, size_t ws_size,
                              hipStream_t stream) {
    const float* x        = (const float*)d_in[0];
    const int*   esrc     = (const int*)d_in[1];
    const int*   edst     = (const int*)d_in[2];
    const float* ew       = (const float*)d_in[3];
    const float* t_p      = (const float*)d_in[4];
    const float* Wg       = (const float*)d_in[5];
    const float* bg       = (const float*)d_in[6];
    const float* proto    = (const float*)d_in[7];
    const float* bp       = (const float*)d_in[8];
    const float* g1       = (const float*)d_in[9];
    const float* b1       = (const float*)d_in[10];
    const float* g2       = (const float*)d_in[11];
    const float* b2       = (const float*)d_in[12];

    float* h_out = (float*)d_out;                 // N x DH
    float* z_out = h_out + (size_t)NN * DH;       // N x DOUT

    // workspace carve-up
    char* ws = (char*)d_ws;
    size_t off = 0;
    float* cur1   = (float*)(ws + off); off += (size_t)NN * DIN * sizeof(float);     // 51.2 MB
    __bf16* xmb   = (__bf16*)(ws + off); off += (size_t)NN * DIN * sizeof(__bf16);   // 25.6 MB
    __bf16* wpack = (__bf16*)(ws + off); off += (size_t)DIN * DH * sizeof(__bf16);
    __bf16* ppack = (__bf16*)(ws + off); off += (size_t)DH * DOUT * sizeof(__bf16);
    float* scale1 = (float*)(ws + off); off += DIN * sizeof(float);
    float* shift1 = (float*)(ws + off); off += DIN * sizeof(float);
    float* scale2 = (float*)(ws + off); off += DH * sizeof(float);
    float* shift2 = (float*)(ws + off); off += DH * sizeof(float);

    float* cur2 = h_out;   // reuse h region of d_out as hop-2 scratch (dead before GEMM1 writes h)

    // 1) zero scatter targets
    k_zero<<<2048, 256, 0, stream>>>(cur1, (long)NN * DIN);
    k_zero<<<2048, 256, 0, stream>>>(cur2, (long)NN * DIN);

    // 2) pack weights into bf16 WMMA B-fragment layouts
    k_pack_w<<<(DIN * DH) / 256, 256, 0, stream>>>(Wg, wpack);
    k_pack_p<<<(DH * DOUT) / 256, 256, 0, stream>>>(proto, ppack);

    // 3) two SPMM hops
    k_spmm<<<EE, DIN, 0, stream>>>(esrc, edst, ew, x, cur1);
    k_spmm<<<EE, DIN, 0, stream>>>(esrc, edst, ew, cur1, cur2);

    // 4) BN1 stats -> affine, fused bf16 conversion
    k_bn1_stats<<<DIN, 256, 0, stream>>>(x, cur1, cur2, g1, b1, scale1, shift1);
    k_xm_bf16<<<4096, 256, 0, stream>>>(x, cur1, cur2, scale1, shift1, xmb);

    // 5) GEMM1 (WMMA bf16, 4 N-tiles per wave): h = relu(xm @ W + b)
    int waves1 = (NN / 16) * ((DH / 16) / 4);            // 12500
    k_gemm1<<<(waves1 + 7) / 8, 256, 0, stream>>>(xmb, wpack, bg, h_out);

    // 6) BN2 stats over h
    k_bn2_stats<<<DH, 256, 0, stream>>>(h_out, g2, b2, scale2, shift2);

    // 7) GEMM2 (WMMA bf16, full 16x64 strip per wave): z_pre = bn2(h) @ P^T + bias_pre
    int waves2 = NN / 16;                                // 3125
    k_gemm2<<<(waves2 + 7) / 8, 256, 0, stream>>>(h_out, ppack, scale2, shift2, bp, z_out);

    // 8) log_softmax(z / t_p) in place
    k_logsoftmax<<<NN, DOUT, 0, stream>>>(z_out, t_p);

    (void)in_sizes; (void)n_in; (void)out_size; (void)ws_size;
}
